// Phraselize_65824668778891
// MI455X (gfx1250) — matrix operand order repaired and checked by reading, not compiled, not run on hardware
//
#include <hip/hip_runtime.h>
#include <hip/hip_bf16.h>
#include <stdint.h>

// Problem constants (from reference): B=8, L=2048, D=1024, P=4
#define B_      8
#define L_      2048
#define D_      1024
#define P_      4
#define M_TOT   (B_ * L_)   // 16384
#define N_TOT   D_          // 1024

typedef __attribute__((ext_vector_type(16))) __bf16   bf16x16;
typedef __attribute__((ext_vector_type(8)))  float    f32x8;
typedef __attribute__((ext_vector_type(4)))  float    f32x4;
typedef __attribute__((ext_vector_type(4)))  unsigned u32x4;
typedef __attribute__((ext_vector_type(2)))  unsigned u32x2;

static __device__ __forceinline__ unsigned short f32_to_bf16_rne(float f) {
    unsigned u = __builtin_bit_cast(unsigned, f);
    unsigned r = u + 0x7FFFu + ((u >> 16) & 1u);   // round-to-nearest-even
    return (unsigned short)(r >> 16);
}
static __device__ __forceinline__ unsigned pk_bf16(float a, float b) {
    return (unsigned)f32_to_bf16_rne(a) | ((unsigned)f32_to_bf16_rne(b) << 16);
}

union FragCast { u32x4 u[2]; bf16x16 v; };

// ---------------------------------------------------------------------------
// One-shot: convert x [B*L, D] f32 -> bf16 xp (same layout). 8 elems/thread.
// ---------------------------------------------------------------------------
__global__ __launch_bounds__(256)
void convert_x_kernel(const float* __restrict__ x, unsigned short* __restrict__ xp) {
    const size_t idx = ((size_t)blockIdx.x * 256 + threadIdx.x) * 8;
    const f32x4 a = *reinterpret_cast<const f32x4*>(x + idx);
    const f32x4 b = *reinterpret_cast<const f32x4*>(x + idx + 4);
    u32x4 o;
    o.x = pk_bf16(a.x, a.y);
    o.y = pk_bf16(a.z, a.w);
    o.z = pk_bf16(b.x, b.y);
    o.w = pk_bf16(b.z, b.w);
    *reinterpret_cast<u32x4*>(xp + idx) = o;
}

// ---------------------------------------------------------------------------
// One-shot: repack W[d, d'*4+i] (f32) into bf16 Wp[i][d'][d] (K-major per
// phase) so GEMM B-tile staging is a pure contiguous copy.
// ---------------------------------------------------------------------------
__global__ __launch_bounds__(256)
void repack_W_kernel(const float* __restrict__ W, unsigned short* __restrict__ Wp) {
    __shared__ __align__(16) unsigned short t[P_][32][40];   // [i][k][n], padded
    const int tid = threadIdx.x;
    const int n0  = blockIdx.x * 32;   // output-feature (d) tile
    const int k0  = blockIdx.y * 32;   // d' tile

#pragma unroll
    for (int s = 0; s < 4; ++s) {
        int c  = tid + (s << 8);       // 0..1023
        int tk = c & 31;
        int tn = c >> 5;
        const f32x4 v = *reinterpret_cast<const f32x4*>(
            W + (size_t)(n0 + tn) * (D_ * P_) + (size_t)(k0 + tk) * P_);
        t[0][tk][tn] = f32_to_bf16_rne(v.x);
        t[1][tk][tn] = f32_to_bf16_rne(v.y);
        t[2][tk][tn] = f32_to_bf16_rne(v.z);
        t[3][tk][tn] = f32_to_bf16_rne(v.w);
    }
    __syncthreads();

#pragma unroll
    for (int s = 0; s < 2; ++s) {
        int c  = tid + (s << 8);       // 0..511
        int i  = c >> 7;               // 0..3
        int k  = (c >> 2) & 31;
        int nq = c & 3;
        u32x4 d = *reinterpret_cast<const u32x4*>(&t[i][k][nq * 8]);
        *reinterpret_cast<u32x4*>(
            Wp + ((size_t)(i * D_ + k0 + k) * D_) + n0 + nq * 8) = d;
    }
}

// ---------------------------------------------------------------------------
// y[m][n] = sum_{phase,k} x[b, l-phase, k] * W[n, k*4+phase] + bias[n]
// Block tile 128x128, BK=32, double-buffered LDS; 8 waves each own 32x64
// (2x4 v_wmma_f32_16x16x32_bf16 accumulators).
// MODE: 0 = x f32 + W f32 raw; 1 = x f32 + W packed bf16; 2 = both packed.
// ---------------------------------------------------------------------------
template <int MODE>
__global__ __launch_bounds__(256)
void phrase_gemm_kernel(const void* __restrict__ xsrc,
                        const void* __restrict__ Wsrc,
                        const float* __restrict__ bias,
                        float* __restrict__ y) {
    __shared__ __align__(16) unsigned short As[2][128][40];   // [buf][m][k] +8 pad
    __shared__ __align__(16) unsigned short Bs[2][32][136];   // [buf][k][n] +8 pad

    const int tid  = threadIdx.x;
    const int lane = tid & 31;
    const int wave = tid >> 5;
    const int wm   = wave >> 1;        // 0..3 : wave row
    const int wn   = wave & 1;         // 0..1 : wave col
    const int g    = lane >> 4;        // half-wave
    const int lm   = lane & 15;

    const int block_m = blockIdx.x * 128;
    const int block_n = blockIdx.y * 128;
    const int batch   = block_m >> 11;            // / L_
    const int l_base  = block_m & (L_ - 1);

    const float* __restrict__ xb_f =
        (const float*)xsrc + (size_t)batch * L_ * D_;
    const unsigned short* __restrict__ xb_h =
        (const unsigned short*)xsrc + (size_t)batch * L_ * D_;

    f32x8 acc[2][4];
#pragma unroll
    for (int i = 0; i < 2; ++i)
#pragma unroll
        for (int j = 0; j < 4; ++j)
#pragma unroll
            for (int r = 0; r < 8; ++r) acc[i][j][r] = 0.0f;

    // ---- A staging: 128x32 tile, shifted by phase, zero-fill l<0 ----
    auto stageA = [&](int buf, int it) {
        const int phase = it >> 5;
        const int k0    = (it & 31) << 5;
        if constexpr (MODE == 2) {
            // pure bf16 copy: 512 x 16B chunks, 2 per thread
#pragma unroll
            for (int s = 0; s < 2; ++s) {
                int c = tid + (s << 8);    // 0..511
                int m = c >> 2;
                int q = c & 3;
                int l = l_base + m - phase;
                u32x4 v = {0u, 0u, 0u, 0u};
                if (l >= 0)
                    v = *reinterpret_cast<const u32x4*>(
                        xb_h + (size_t)l * D_ + k0 + (q << 3));
                *reinterpret_cast<u32x4*>(&As[buf][m][q << 3]) = v;
            }
        } else {
            // f32 load + inline convert: 1024 float4 chunks, 4 per thread
#pragma unroll
            for (int s = 0; s < 4; ++s) {
                int c  = tid + (s << 8);
                int m  = c >> 3;
                int kq = c & 7;
                int l  = l_base + m - phase;
                unsigned lo = 0u, hi = 0u;
                if (l >= 0) {
                    const f32x4 v = *reinterpret_cast<const f32x4*>(
                        xb_f + (size_t)l * D_ + k0 + (kq << 2));
                    lo = pk_bf16(v.x, v.y);
                    hi = pk_bf16(v.z, v.w);
                }
                u32x2 pk; pk.x = lo; pk.y = hi;
                *reinterpret_cast<u32x2*>(&As[buf][m][kq << 2]) = pk;
            }
        }
    };

    // ---- B staging: 32(k) x 128(n) bf16 tile, K-major in LDS ----
    auto stageB = [&](int buf, int it) {
        const int phase = it >> 5;
        const int k0    = (it & 31) << 5;
        if constexpr (MODE >= 1) {
            const unsigned short* Wp = (const unsigned short*)Wsrc;
#pragma unroll
            for (int s = 0; s < 2; ++s) {
                int c = tid + (s << 8);    // 0..511 16B chunks
                int k = c >> 4;
                int q = c & 15;
                const u32x4 v = *reinterpret_cast<const u32x4*>(
                    Wp + ((size_t)(phase * D_ + k0 + k) * D_) + block_n + (q << 3));
                *reinterpret_cast<u32x4*>(&Bs[buf][k][q << 3]) = v;
            }
        } else {
            const float* W = (const float*)Wsrc;
#pragma unroll
            for (int s = 0; s < 16; ++s) {
                int c = tid + (s << 8);
                int k = c & 31;
                int n = c >> 5;
                float w = W[(size_t)(block_n + n) * (D_ * P_) +
                            (size_t)(k0 + k) * P_ + phase];
                Bs[buf][k][n] = f32_to_bf16_rne(w);
            }
        }
    };

    // ---- fragments + WMMA ----
    auto compute = [&](int buf) {
        bf16x16 aF[2], bF[4];
#pragma unroll
        for (int mt = 0; mt < 2; ++mt) {
            // A 16x32 bf16 frag: lane = M row; two contiguous 16B K-runs
            const int row = wm * 32 + mt * 16 + lm;
            FragCast f;
            f.u[0] = *reinterpret_cast<const u32x4*>(&As[buf][row][g << 3]);
            f.u[1] = *reinterpret_cast<const u32x4*>(&As[buf][row][16 + (g << 3)]);
            aF[mt] = f.v;
        }
#pragma unroll
        for (int nt = 0; nt < 4; ++nt) {
            // B 32x16 bf16 frag: lane = N col; K = 8g+s (s<8), 16+8g+(s-8) (s>=8)
            const int ncol = wn * 64 + nt * 16;
            struct { unsigned short e[16]; } f;
#pragma unroll
            for (int s = 0; s < 16; ++s)
                f.e[s] = Bs[buf][(s < 8 ? (g << 3) + s : 16 + (g << 3) + (s - 8))][ncol + lm];
            bF[nt] = __builtin_bit_cast(bf16x16, f);
        }
#pragma unroll
        for (int mt = 0; mt < 2; ++mt)
#pragma unroll
            for (int nt = 0; nt < 4; ++nt)
                acc[mt][nt] = __builtin_amdgcn_wmma_f32_16x16x32_bf16(
                    false, aF[mt], false, bF[nt], (short)0, acc[mt][nt],
                    false, false);
    };

    // ---- main loop: 4 phases x 32 K-chunks = 128 iterations, double-buffered
    stageA(0, 0);
    stageB(0, 0);
    __syncthreads();
    int buf = 0;
    for (int it = 0; it < 4 * (D_ / 32); ++it) {
        if (it + 1 < 4 * (D_ / 32)) {
            stageA(buf ^ 1, it + 1);
            stageB(buf ^ 1, it + 1);
        }
        compute(buf);
        __syncthreads();
        buf ^= 1;
    }

    // ---- epilogue: C 16x16 f32 layout: lane -> N=lm, VGPR r -> M=r+8*g ----
#pragma unroll
    for (int nt = 0; nt < 4; ++nt) {
        const int n  = block_n + wn * 64 + nt * 16 + lm;
        const float bv = bias[n];
#pragma unroll
        for (int mt = 0; mt < 2; ++mt) {
            const int m0 = block_m + wm * 32 + mt * 16 + (g << 3);
#pragma unroll
            for (int r = 0; r < 8; ++r) {
                y[(size_t)(m0 + r) * N_TOT + n] = acc[mt][nt][r] + bv;
            }
        }
    }
}

// ---------------------------------------------------------------------------
extern "C" void kernel_launch(void* const* d_in, const int* in_sizes, int n_in,
                              void* d_out, int out_size, void* d_ws, size_t ws_size,
                              hipStream_t stream) {
    const float* x    = (const float*)d_in[0];   // [B, L, D] f32
    const float* W    = (const float*)d_in[1];   // [D, D*P] f32
    const float* bias = (const float*)d_in[2];   // [D] f32
    float* y          = (float*)d_out;           // [B, L, D] f32
    (void)in_sizes; (void)n_in; (void)out_size;

    const size_t wp_bytes = (size_t)P_ * D_ * D_ * sizeof(unsigned short);   //  8 MiB
    const size_t xp_bytes = (size_t)M_TOT * D_ * sizeof(unsigned short);     // 32 MiB
    dim3 gemm_grid(M_TOT / 128, N_TOT / 128, 1);   // (128, 8)

    if (ws_size >= wp_bytes + xp_bytes) {
        unsigned short* Wp = (unsigned short*)d_ws;
        unsigned short* xp = (unsigned short*)((char*)d_ws + wp_bytes);
        repack_W_kernel<<<dim3(D_ / 32, D_ / 32, 1), 256, 0, stream>>>(W, Wp);
        convert_x_kernel<<<dim3((M_TOT * D_) / (256 * 8), 1, 1), 256, 0, stream>>>(x, xp);
        phrase_gemm_kernel<2><<<gemm_grid, 256, 0, stream>>>(
            (const void*)xp, (const void*)Wp, bias, y);
    } else if (ws_size >= wp_bytes) {
        unsigned short* Wp = (unsigned short*)d_ws;
        repack_W_kernel<<<dim3(D_ / 32, D_ / 32, 1), 256, 0, stream>>>(W, Wp);
        phrase_gemm_kernel<1><<<gemm_grid, 256, 0, stream>>>(
            (const void*)x, (const void*)Wp, bias, y);
    } else {
        phrase_gemm_kernel<0><<<gemm_grid, 256, 0, stream>>>(
            (const void*)x, (const void*)W, bias, y);
    }
}
